// GINBert_32160715112912
// MI455X (gfx1250) — matrix operand order repaired and checked by reading, not compiled, not run on hardware
//
#include <hip/hip_runtime.h>
#include <stdint.h>

// ---------------------------------------------------------------------------
// GINBert forward for MI455X (gfx1250, wave32).
// Heavy compute (protein MLP, ~189 GFLOP) runs on v_wmma_f32_16x16x32_bf16.
// Weights are pre-converted to bf16 (transposed) once per launch so the GEMM
// inner loop is pure b128 copies + ds_load_b128 fragment loads + WMMA bursts.
// ---------------------------------------------------------------------------

#define NB    256     // batch
#define NPG   30      // nodes per graph
#define NN    7680    // total nodes
#define NE    16384   // edges
#define FIN   78
#define DD    32
#define MAXN  45
#define LPP   512     // protein length
#define PEE   1280    // protein embed dim

typedef __bf16 bf16;
typedef bf16  v16bf __attribute__((ext_vector_type(16)));
typedef float v8f   __attribute__((ext_vector_type(8)));

union Frag { v16bf bf; uint4 q[2]; };

static __device__ __forceinline__ unsigned short f2bf(float f) {
    union { float f; unsigned u; } v; v.f = f;
    unsigned r = v.u + 0x7FFFu + ((v.u >> 16) & 1u);   // round-to-nearest-even
    return (unsigned short)(r >> 16);
}
static __device__ __forceinline__ unsigned pack2bf(float a, float b) {
    return (unsigned)f2bf(a) | ((unsigned)f2bf(b) << 16);
}
static __device__ __forceinline__ v8f vzero8() {
    v8f z;
#pragma unroll
    for (int i = 0; i < 8; ++i) z[i] = 0.f;
    return z;
}

// ---------------------------------------------------------------------------
// One-time: dst[n][k] (bf16) = src[k][n] (f32).  (transpose + convert)
// ---------------------------------------------------------------------------
__global__ void cvt_transpose_bf16(const float* __restrict__ src,
                                   unsigned short* __restrict__ dst,
                                   int K, int N) {
    int idx = blockIdx.x * blockDim.x + threadIdx.x;
    if (idx >= K * N) return;
    int n = idx / K, k = idx - n * K;
    dst[(size_t)n * K + k] = f2bf(src[(size_t)k * N + n]);
}

// ---------------------------------------------------------------------------
// GIN: edge scatter-add  agg[dst,:] += h[src,:]
// ---------------------------------------------------------------------------
__global__ void gin_scatter(const float* __restrict__ h, const int* __restrict__ ei,
                            float* __restrict__ agg, int K) {
    int idx = blockIdx.x * blockDim.x + threadIdx.x;
    if (idx >= NE * K) return;
    int e = idx / K, k = idx - e * K;
    int src = ei[e], dst = ei[NE + e];
    atomicAdd(&agg[(size_t)dst * K + k], h[(size_t)src * K + k]);
}

// ---------------------------------------------------------------------------
// GIN: per-graph MLP (relu(relu((h+agg)W1+b1)W2+b2)) + GraphNorm.
// One wave (32 lanes, lane = feature) per graph; 30 contiguous nodes.
// ---------------------------------------------------------------------------
__global__ __launch_bounds__(32) void gin_mlp_norm(
    const float* __restrict__ hin, const float* __restrict__ agg, int K,
    const float* __restrict__ w1, const float* __restrict__ b1,
    const float* __restrict__ w2, const float* __restrict__ b2,
    const float* __restrict__ gg, const float* __restrict__ gb,
    const float* __restrict__ ga, float* __restrict__ hout) {
    __shared__ float inr[96];
    __shared__ float tr[DD];
    __shared__ float h2b[NPG][DD];
    const int j = threadIdx.x;
    const int base = blockIdx.x * NPG;

    for (int i = 0; i < NPG; ++i) {
        const int node = base + i;
        for (int k = j; k < K; k += 32)
            inr[k] = hin[(size_t)node * K + k] + agg[(size_t)node * K + k];
        __syncthreads();
        float a1 = b1[j];
        for (int k = 0; k < K; ++k) a1 += inr[k] * w1[k * DD + j];
        tr[j] = fmaxf(a1, 0.f);
        __syncthreads();
        float a2 = b2[j];
#pragma unroll
        for (int k = 0; k < DD; ++k) a2 += tr[k] * w2[k * DD + j];
        h2b[i][j] = fmaxf(a2, 0.f);
        __syncthreads();
    }
    float mean = 0.f;
    for (int i = 0; i < NPG; ++i) mean += h2b[i][j];
    mean *= (1.f / NPG);
    const float aj = ga[j];
    float var = 0.f;
    for (int i = 0; i < NPG; ++i) { float o = h2b[i][j] - aj * mean; var += o * o; }
    var *= (1.f / NPG);
    const float inv = rsqrtf(var + 1e-5f);
    const float gj = gg[j], bj = gb[j];
    for (int i = 0; i < NPG; ++i) {
        float o = (h2b[i][j] - aj * mean) * inv;
        hout[(size_t)(base + i) * DD + j] = gj * o + bj;
    }
}

// ---------------------------------------------------------------------------
// xd[b,i,:] = relu(dense_row @ fc1_w + fc1_b);  one wave per (b,i) row.
// ---------------------------------------------------------------------------
__global__ __launch_bounds__(256) void fc1_kernel(
    const float* __restrict__ h, const float* __restrict__ w,
    const float* __restrict__ bi, float* __restrict__ xd) {
    __shared__ float rowb[8][DD];
    const int tid = threadIdx.x, wv = tid >> 5, lane = tid & 31;
    const int widx = blockIdx.x * 8 + wv;          // widx < 256*45
    const int b = widx / MAXN, i = widx - b * MAXN;
    rowb[wv][lane] = (i < NPG) ? h[((size_t)b * NPG + i) * DD + lane] : 0.f;
    __syncthreads();
    float* o = xd + (size_t)widx * 128;
#pragma unroll
    for (int q = 0; q < 4; ++q) {
        const int jj = lane + q * 32;
        float acc = bi[jj];
#pragma unroll
        for (int k = 0; k < DD; ++k) acc += rowb[wv][k] * w[k * 128 + jj];
        o[jj] = fmaxf(acc, 0.f);
    }
}

// ---------------------------------------------------------------------------
// Fused protein MLP:  t2 = relu(relu(T @ W1 + b1) @ W2 + b2)
//   T   = proteins[target_id[b]]   : [512,1280] f32 (converted to bf16 on stage)
//   w1t : [512][1280] bf16 (pre-transposed W1), w2t : [128][512] bf16
// One block per (batch, 32-row M tile). The 32x512 bf16 intermediate lives
// entirely in LDS; both phases use v_wmma_f32_16x16x32_bf16.
//
// ISA fragment layouts (CDNA5 ISA §7.12.2):
//   A 16x32 bf16 : lane l -> row = l&15; K runs [kh*8,+8) and [16+kh*8,+8),
//                  kh = l>>4  (two 16B ds loads)
//   B 32x16 bf16 : lane l -> col = l&15; K runs [kh*8,+8) and [16+kh*8,+8) as A
//                  (two 16B ds loads from a transposed [N][K] LDS tile)
//   C/D 16x16 f32: vgpr r, lane l -> row = r + 8*(l>>4), col = l&15
// ---------------------------------------------------------------------------
__global__ __launch_bounds__(256) void pf_fused(
    const float* __restrict__ proteins, const int* __restrict__ target_id,
    const unsigned short* __restrict__ w1t, const float* __restrict__ b1,
    const unsigned short* __restrict__ w2t, const float* __restrict__ b2,
    float* __restrict__ t2out) {
    // LDS carve: A k-tile [32][40], W k-tile transposed [512][40], T1 [32][520]
    __shared__ alignas(16) unsigned short lds[32 * 40 + 512 * 40 + 32 * 520];
    unsigned short* Ast = lds;                       // [32][40]
    unsigned short* Wst = lds + 32 * 40;             // [512][40] (ph1) / [128][40] (ph2)
    unsigned short* T1  = lds + 32 * 40 + 512 * 40;  // [32][520]

    const int tid  = threadIdx.x;
    const int w    = tid >> 5, lane = tid & 31;
    const int b    = blockIdx.x >> 4;
    const int m0   = (blockIdx.x & 15) * 32;
    const int pid  = target_id[b];
    const float* A = proteins + (size_t)pid * LPP * PEE + (size_t)m0 * PEE;

    const int sm = w & 1;        // M subtile (0/1)
    const int ng = w >> 1;       // N group (0..3), 128 cols each
    const int lrow = lane & 15, lhalf = lane >> 4;

    // ---------------- phase 1: T1 = relu(A @ W1 + b1), N = 512 ----------------
    v8f acc[8];
#pragma unroll
    for (int j = 0; j < 8; ++j) acc[j] = vzero8();

    for (int kt = 0; kt < PEE / 32; ++kt) {
        const int k0 = kt * 32;
        {   // stage A tile (f32 -> packed bf16): one float4 -> one b64 store
            const int r = tid >> 3, c4 = (tid & 7) * 4;
            float4 v = *(const float4*)(A + (size_t)r * PEE + k0 + c4);
            uint2 p;
            p.x = pack2bf(v.x, v.y);
            p.y = pack2bf(v.z, v.w);
            *(uint2*)(Ast + r * 40 + c4) = p;
        }
        // stage W1 k-tile: pure b128 copy from pre-transposed bf16 weights
#pragma unroll
        for (int u = tid; u < 2048; u += 256) {        // 512 n x 4 (8-short units)
            const int n = u >> 2, k8 = (u & 3) * 8;
            *(uint4*)(Wst + n * 40 + k8) =
                *(const uint4*)(w1t + (size_t)n * PEE + k0 + k8);
        }
        __syncthreads();

        Frag a;
        a.q[0] = *(const uint4*)(Ast + (sm * 16 + lrow) * 40 + lhalf * 8);
        a.q[1] = *(const uint4*)(Ast + (sm * 16 + lrow) * 40 + 16 + lhalf * 8);
        Frag bb[8];
#pragma unroll
        for (int j = 0; j < 8; ++j) {
            const uint4* bp = (const uint4*)(Wst + (ng * 128 + j * 16 + lrow) * 40 + lhalf * 8);
            bb[j].q[0] = bp[0]; bb[j].q[1] = bp[2];
        }
#pragma unroll
        for (int j = 0; j < 8; ++j)
            acc[j] = __builtin_amdgcn_wmma_f32_16x16x32_bf16(
                false, a.bf, false, bb[j].bf, (short)0, acc[j], false, false);
        __syncthreads();
    }
    // epilogue -> bf16 tile in LDS
#pragma unroll
    for (int j = 0; j < 8; ++j) {
        const int col = ng * 128 + j * 16 + lrow;
        const float bias = b1[col];
#pragma unroll
        for (int r = 0; r < 8; ++r) {
            const int row = sm * 16 + r + lhalf * 8;
            float v = acc[j][r] + bias;
            T1[row * 520 + col] = f2bf(v > 0.f ? v : 0.f);
        }
    }
    __syncthreads();

    // ---------------- phase 2: t2 = relu(T1 @ W2 + b2), N = 128 ----------------
    v8f acc2[2];
    acc2[0] = vzero8(); acc2[1] = vzero8();
    const int sm2 = w & 1;
    const int nh  = w >> 1;     // n subtiles nh*2, nh*2+1

    for (int kt = 0; kt < LPP / 32; ++kt) {
        const int k0 = kt * 32;
#pragma unroll
        for (int u = tid; u < 512; u += 256) {         // 128 n x 4 (8-short units)
            const int n = u >> 2, k8 = (u & 3) * 8;
            *(uint4*)(Wst + n * 40 + k8) =
                *(const uint4*)(w2t + (size_t)n * LPP + k0 + k8);
        }
        __syncthreads();

        Frag a;
        a.q[0] = *(const uint4*)(T1 + (sm2 * 16 + lrow) * 520 + k0 + lhalf * 8);
        a.q[1] = *(const uint4*)(T1 + (sm2 * 16 + lrow) * 520 + k0 + 16 + lhalf * 8);
        Frag bb[2];
#pragma unroll
        for (int j = 0; j < 2; ++j) {
            const uint4* bp = (const uint4*)(Wst + ((nh * 2 + j) * 16 + lrow) * 40 + lhalf * 8);
            bb[j].q[0] = bp[0]; bb[j].q[1] = bp[2];
        }
#pragma unroll
        for (int j = 0; j < 2; ++j)
            acc2[j] = __builtin_amdgcn_wmma_f32_16x16x32_bf16(
                false, a.bf, false, bb[j].bf, (short)0, acc2[j], false, false);
        __syncthreads();
    }
#pragma unroll
    for (int j = 0; j < 2; ++j) {
        const int col = (nh * 2 + j) * 16 + lrow;
        const float bias = b2[col];
#pragma unroll
        for (int r = 0; r < 8; ++r) {
            const int row = m0 + sm2 * 16 + r + lhalf * 8;
            float v = acc2[j][r] + bias;
            t2out[((size_t)b * LPP + row) * 128 + col] = v > 0.f ? v : 0.f;
        }
    }
}

// ---------------------------------------------------------------------------
// Co-attention + pooling + final MLP head. One block per batch element.
// ---------------------------------------------------------------------------
__global__ __launch_bounds__(256) void coatt(
    const float* __restrict__ xd, const float* __restrict__ t2,
    const float* __restrict__ Wb, const float* __restrict__ Wc,
    const float* __restrict__ Wp, const float* __restrict__ whc,
    const float* __restrict__ whp,
    const float* __restrict__ cf1w, const float* __restrict__ cf1b,
    const float* __restrict__ cf2w, const float* __restrict__ cf2b,
    const float* __restrict__ outw, const float* __restrict__ outb,
    float* __restrict__ out) {
    __shared__ float S[23600];
    float* xd_s  = S;              // [45][128]
    float* tmp_s = xd_s + 5760;    // [128][48]   (later reused as hcv)
    float* wcx   = tmp_s + 6144;   // [64][48]
    float* hcacc = wcx + 3072;     // [64][48]
    float* t2c   = hcacc + 3072;   // [16][128]   (later reused as hpv [64][16])
    float* cch   = t2c + 2048;     // [16][48]
    float* wpt   = cch + 768;      // [64][16]
    float* ps    = wpt + 1024;     // [512]
    float* cs    = ps + 512;       // [48]
    float* red   = cs + 48;        // [256]
    float* comp  = red + 256;      // [128]
    float* prot  = comp + 128;     // [128]
    float* h1    = prot + 128;     // [512]
    float* h2    = h1 + 512;       // [128]

    const int tid = threadIdx.x;
    const int b = blockIdx.x;
    const float* xdb = xd + (size_t)b * MAXN * 128;
    const float* t2b = t2 + (size_t)b * LPP * 128;

    for (int idx = tid; idx < MAXN * 128; idx += 256) xd_s[idx] = xdb[idx];
    __syncthreads();

    // tmp[d][s] = dot(Wb[d,:], xd[s,:])
    for (int idx = tid; idx < 128 * MAXN; idx += 256) {
        const int d = idx / MAXN, s = idx - d * MAXN;
        const float* wbr = Wb + d * 128;
        float acc = 0.f;
        for (int e = 0; e < 128; ++e) acc += wbr[e] * xd_s[s * 128 + e];
        tmp_s[d * 48 + s] = acc;
    }
    // wcx[c][s] = dot(Wc[c,:], xd[s,:])
    for (int idx = tid; idx < 64 * MAXN; idx += 256) {
        const int c = idx / MAXN, s = idx - c * MAXN;
        const float* wcr = Wc + c * 128;
        float acc = 0.f;
        for (int e = 0; e < 128; ++e) acc += wcr[e] * xd_s[s * 128 + e];
        wcx[c * 48 + s] = acc;
    }
    for (int idx = tid; idx < 64 * 48; idx += 256) hcacc[idx] = 0.f;
    __syncthreads();

    float* hpv = t2c;   // alias, safe: t2c fully consumed before hpv writes
    for (int lc = 0; lc < LPP / 16; ++lc) {
        const int l0 = lc * 16;
        for (int idx = tid; idx < 16 * 128; idx += 256) t2c[idx] = t2b[l0 * 128 + idx];
        __syncthreads();
        // C chunk: cch[i][s] = tanh(dot(t2[l0+i,:], tmp[:,s]))
        for (int idx = tid; idx < 16 * MAXN; idx += 256) {
            const int i = idx / MAXN, s = idx - i * MAXN;
            float acc = 0.f;
            for (int d = 0; d < 128; ++d) acc += t2c[i * 128 + d] * tmp_s[d * 48 + s];
            cch[i * 48 + s] = tanhf(acc);
        }
        // wpt[c][i] = dot(Wp[c,:], t2[l0+i,:])
        for (int idx = tid; idx < 1024; idx += 256) {
            const int c = idx >> 4, i = idx & 15;
            const float* wpr = Wp + c * 128;
            float acc = 0.f;
            for (int d = 0; d < 128; ++d) acc += wpr[d] * t2c[i * 128 + d];
            wpt[idx] = acc;
        }
        __syncthreads();
        // Hc accumulator: hcacc[c][s] += sum_i wpt[c][i]*cch[i][s]
        for (int idx = tid; idx < 64 * MAXN; idx += 256) {
            const int c = idx / MAXN, s = idx - c * MAXN;
            float acc = 0.f;
#pragma unroll
            for (int i = 0; i < 16; ++i) acc += wpt[c * 16 + i] * cch[i * 48 + s];
            hcacc[c * 48 + s] += acc;
        }
        // Hp scores: hpv[c][i] = whp[c]*tanh(wpt[c][i] + sum_s wcx[c][s]*cch[i][s])
        for (int idx = tid; idx < 1024; idx += 256) {
            const int c = idx >> 4, i = idx & 15;
            float acc = wpt[idx];
            for (int s = 0; s < MAXN; ++s) acc += wcx[c * 48 + s] * cch[i * 48 + s];
            hpv[c * 16 + i] = whp[c] * tanhf(acc);
        }
        __syncthreads();
        if (tid < 16) {
            float acc = 0.f;
            for (int c = 0; c < 64; ++c) acc += hpv[c * 16 + tid];
            ps[l0 + tid] = acc;
        }
        __syncthreads();
    }

    // Hc scores
    float* hcv = tmp_s;   // reuse
    for (int idx = tid; idx < 64 * MAXN; idx += 256) {
        const int c = idx / MAXN, s = idx - c * MAXN;
        hcv[c * 48 + s] = whc[c] * tanhf(wcx[c * 48 + s] + hcacc[c * 48 + s]);
    }
    __syncthreads();
    if (tid < MAXN) {
        float acc = 0.f;
        for (int c = 0; c < 64; ++c) acc += hcv[c * 48 + tid];
        cs[tid] = acc;
    }
    __syncthreads();
    // softmax over s (45)
    if (tid == 0) {
        float m = cs[0];
        for (int s = 1; s < MAXN; ++s) m = fmaxf(m, cs[s]);
        red[0] = m;
    }
    __syncthreads();
    if (tid < MAXN) cs[tid] = __expf(cs[tid] - red[0]);
    __syncthreads();
    if (tid == 0) {
        float su = 0.f;
        for (int s = 0; s < MAXN; ++s) su += cs[s];
        red[1] = 1.f / su;
    }
    __syncthreads();
    if (tid < MAXN) cs[tid] *= red[1];
    __syncthreads();
    // softmax over l (512), parallel
    {
        red[tid] = fmaxf(ps[tid], ps[tid + 256]);
        __syncthreads();
        for (int off = 128; off > 0; off >>= 1) {
            if (tid < off) red[tid] = fmaxf(red[tid], red[tid + off]);
            __syncthreads();
        }
        const float mx = red[0];
        __syncthreads();
        const float e0 = __expf(ps[tid] - mx), e1 = __expf(ps[tid + 256] - mx);
        ps[tid] = e0; ps[tid + 256] = e1;
        red[tid] = e0 + e1;
        __syncthreads();
        for (int off = 128; off > 0; off >>= 1) {
            if (tid < off) red[tid] += red[tid + off];
            __syncthreads();
        }
        const float inv = 1.f / red[0];
        __syncthreads();
        ps[tid] *= inv; ps[tid + 256] *= inv;
    }
    __syncthreads();
    // pooled features
    if (tid < 128) {
        float acc = 0.f;
        for (int s = 0; s < MAXN; ++s) acc += cs[s] * xd_s[s * 128 + tid];
        comp[tid] = acc;
        float pacc = 0.f;
        for (int l = 0; l < LPP; ++l) pacc += ps[l] * t2b[l * 128 + tid];
        prot[tid] = pacc;
    }
    __syncthreads();
    // head: relu(cp@cf1+b) -> relu(@cf2+b) -> @out_w + out_b
    for (int o = tid; o < 512; o += 256) {
        float acc = cf1b[o];
        for (int k = 0; k < 128; ++k) acc += comp[k] * cf1w[k * 512 + o];
        for (int k = 0; k < 128; ++k) acc += prot[k] * cf1w[(128 + k) * 512 + o];
        h1[o] = fmaxf(acc, 0.f);
    }
    __syncthreads();
    if (tid < 128) {
        float acc = cf2b[tid];
        for (int k = 0; k < 512; ++k) acc += h1[k] * cf2w[k * 128 + tid];
        h2[tid] = fmaxf(acc, 0.f);
    }
    __syncthreads();
    if (tid < 128) red[tid] = h2[tid] * outw[tid];
    __syncthreads();
    if (tid == 0) {
        float acc = 0.f;
        for (int k = 0; k < 128; ++k) acc += red[k];
        out[b] = acc + outb[0];
    }
}

// ---------------------------------------------------------------------------
extern "C" void kernel_launch(void* const* d_in, const int* in_sizes, int n_in,
                              void* d_out, int out_size, void* d_ws, size_t ws_size,
                              hipStream_t stream) {
    (void)in_sizes; (void)n_in; (void)out_size; (void)ws_size;
    const float* x      = (const float*)d_in[0];
    const float* prots  = (const float*)d_in[1];
    const float* g1_w1  = (const float*)d_in[2];
    const float* g1_b1  = (const float*)d_in[3];
    const float* g1_w2  = (const float*)d_in[4];
    const float* g1_b2  = (const float*)d_in[5];
    const float* gn1_g  = (const float*)d_in[6];
    const float* gn1_b  = (const float*)d_in[7];
    const float* gn1_a  = (const float*)d_in[8];
    const float* gw1s   = (const float*)d_in[9];
    const float* gb1s   = (const float*)d_in[10];
    const float* gw2s   = (const float*)d_in[11];
    const float* gb2s   = (const float*)d_in[12];
    const float* gngs   = (const float*)d_in[13];
    const float* gnbs   = (const float*)d_in[14];
    const float* gnas   = (const float*)d_in[15];
    const float* fc1_w  = (const float*)d_in[16];
    const float* fc1_b  = (const float*)d_in[17];
    const float* pf1_w  = (const float*)d_in[18];
    const float* pf1_b  = (const float*)d_in[19];
    const float* pf2_w  = (const float*)d_in[20];
    const float* pf2_b  = (const float*)d_in[21];
    const float* Wb     = (const float*)d_in[22];
    const float* Wc     = (const float*)d_in[23];
    const float* Wp     = (const float*)d_in[24];
    const float* whc    = (const float*)d_in[25];
    const float* whp    = (const float*)d_in[26];
    const float* cf1_w  = (const float*)d_in[27];
    const float* cf1_b  = (const float*)d_in[28];
    const float* cf2_w  = (const float*)d_in[29];
    const float* cf2_b  = (const float*)d_in[30];
    const float* out_w  = (const float*)d_in[31];
    const float* out_b  = (const float*)d_in[32];
    const int*   ei     = (const int*)d_in[33];
    const int*   tgt    = (const int*)d_in[34];
    // d_in[35] = batch: structure (30 contiguous nodes/graph) is fixed by setup.
    float* out = (float*)d_out;

    // workspace layout (bytes), all offsets 16B-aligned
    char* ws = (char*)d_ws;
    const size_t o_agg = 0;                                    // 2,396,160
    const size_t o_h0  = o_agg + (size_t)NN * FIN * 4;
    const size_t o_h1  = o_h0 + (size_t)NN * DD * 4;           // +983,040
    const size_t o_xd  = o_h1 + (size_t)NN * DD * 4;           // +983,040
    const size_t o_t2  = o_xd + (size_t)NB * MAXN * 128 * 4;   // +5,898,240
    const size_t o_w1t = o_t2 + (size_t)NB * LPP * 128 * 4;    // +67,108,864
    const size_t o_w2t = o_w1t + (size_t)LPP * PEE * 2;        // +1,310,720
    float* agg = (float*)(ws + o_agg);
    float* h0  = (float*)(ws + o_h0);
    float* h1  = (float*)(ws + o_h1);
    float* xd  = (float*)(ws + o_xd);
    float* t2  = (float*)(ws + o_t2);
    unsigned short* w1t = (unsigned short*)(ws + o_w1t);       // [512][1280] bf16
    unsigned short* w2t = (unsigned short*)(ws + o_w2t);       // [128][512]  bf16

    // ---- one-time weight convert/transpose (bf16) ----
    cvt_transpose_bf16<<<(PEE * LPP + 255) / 256, 256, 0, stream>>>(pf1_w, w1t, PEE, LPP);
    cvt_transpose_bf16<<<(LPP * 128 + 255) / 256, 256, 0, stream>>>(pf2_w, w2t, LPP, 128);

    // ---- GIN layer 1 (K = 78) ----
    hipMemsetAsync(agg, 0, (size_t)NN * FIN * 4, stream);
    gin_scatter<<<(NE * FIN + 255) / 256, 256, 0, stream>>>(x, ei, agg, FIN);
    gin_mlp_norm<<<NB, 32, 0, stream>>>(x, agg, FIN, g1_w1, g1_b1, g1_w2, g1_b2,
                                        gn1_g, gn1_b, gn1_a, h0);
    // ---- GIN layers 2..5 (K = 32) ----
    float* hin = h0; float* hou = h1;
    for (int L = 0; L < 4; ++L) {
        hipMemsetAsync(agg, 0, (size_t)NN * DD * 4, stream);
        gin_scatter<<<(NE * DD + 255) / 256, 256, 0, stream>>>(hin, ei, agg, DD);
        gin_mlp_norm<<<NB, 32, 0, stream>>>(hin, agg, DD,
                                            gw1s + (size_t)L * DD * DD, gb1s + (size_t)L * DD,
                                            gw2s + (size_t)L * DD * DD, gb2s + (size_t)L * DD,
                                            gngs + (size_t)L * DD, gnbs + (size_t)L * DD,
                                            gnas + (size_t)L * DD, hou);
        float* t = hou; hou = hin; hin = t;
    }
    // hin now holds the final node features.

    // ---- dense batch + fc1 ----
    fc1_kernel<<<(NB * MAXN) / 8, 256, 0, stream>>>(hin, fc1_w, fc1_b, xd);

    // ---- protein branch: fused pf1+pf2 bf16 WMMA GEMMs ----
    pf_fused<<<NB * (LPP / 32), 256, 0, stream>>>(prots, tgt, w1t, pf1_b,
                                                  w2t, pf2_b, t2);

    // ---- co-attention + head ----
    coatt<<<NB, 256, 0, stream>>>(xd, t2, Wb, Wc, Wp, whc, whp,
                                  cf1_w, cf1_b, cf2_w, cf2_b, out_w, out_b, out);
}
